// NonLocalAttention_61383672594806
// MI455X (gfx1250) — matrix operand order repaired and checked by reading, not compiled, not run on hardware
//
#include <hip/hip_runtime.h>

// ---------------------------------------------------------------------------
// Non-local attention block for MI455X (gfx1250), bf16 WMMA pipeline with
// register-blocked GEMMs; the big o = g @ beta^T GEMM uses a double-buffered
// global->LDS async-copy pipeline (ASYNCcnt) for the shared A panel.
// ---------------------------------------------------------------------------

typedef __attribute__((ext_vector_type(16))) __bf16 v16bf;
typedef __attribute__((ext_vector_type(8)))  __bf16 v8bf;
typedef __attribute__((ext_vector_type(8)))  float  v8f;

static __device__ __forceinline__ __bf16 f2bf(float f) {
  unsigned u = __builtin_bit_cast(unsigned, f);
  u += 0x7FFFu + ((u >> 16) & 1u);            // round-to-nearest-even
  unsigned short s = (unsigned short)(u >> 16);
  return __builtin_bit_cast(__bf16, s);
}
static __device__ __forceinline__ float bf2f(__bf16 h) {
  unsigned short s = __builtin_bit_cast(unsigned short, h);
  unsigned u = ((unsigned)s) << 16;
  return __builtin_bit_cast(float, u);
}

// A fragment (16x32, 16-bit): lane m = lane&15 holds row m.
// lanes 0-15: elems 0..7 = K k0+0..7,  elems 8..15 = K k0+16..23
// lanes16-31: elems 0..7 = K k0+8..15, elems 8..15 = K k0+24..31
static __device__ __forceinline__ v16bf load_frag_a(const __bf16* __restrict__ base,
                                                    int ld, int row0, int k0) {
  int lane = threadIdx.x & 31;
  int m  = lane & 15;
  int hi = (lane >> 4) << 3;                  // 0 or 8
  const __bf16* p = base + (size_t)(row0 + m) * ld + k0 + hi;
  union { v16bf v; v8bf h[2]; } u;
  u.h[0] = *(const v8bf*)(p);
  u.h[1] = *(const v8bf*)(p + 16);
  return u.v;
}

// Same A-fragment gather out of an LDS panel laid out [rows][32] bf16.
static __device__ __forceinline__ v16bf load_frag_a_lds(const __bf16* panel,
                                                        int row0) {
  int lane = threadIdx.x & 31;
  int m  = lane & 15;
  int hi = (lane >> 4) << 3;
  const __bf16* p = panel + (row0 + m) * 32 + hi;
  union { v16bf v; v8bf h[2]; } u;
  u.h[0] = *(const v8bf*)(p);
  u.h[1] = *(const v8bf*)(p + 16);
  return u.v;
}

// B fragment (32x16, 16-bit): lane n = lane&15 holds column n; K order as A fragment.
static __device__ __forceinline__ v16bf load_frag_b(const __bf16* __restrict__ base,
                                                    int ld, int k0, int col0) {
  int lane = threadIdx.x & 31;
  int hi = (lane >> 4) << 3;
  const __bf16* p = base + (size_t)(k0 + hi) * ld + col0 + (lane & 15);
  v16bf v;
#pragma unroll
  for (int s = 0; s < 8; ++s) {
    v[s]     = p[(size_t)s * ld];
    v[s + 8] = p[(size_t)(s + 16) * ld];
  }
  return v;
}

static __device__ __forceinline__ v8f wmma_bf16(v16bf a, v16bf b, v8f c) {
  return __builtin_amdgcn_wmma_f32_16x16x32_bf16(
      /*neg_a=*/false, a, /*neg_b=*/false, b,
      /*c_mod=*/(short)0, c, /*reuse_a=*/false, /*reuse_b=*/false);
}

// ---------------------------------------------------------------------------
__global__ void k_f32_to_bf16(const float* __restrict__ src,
                              __bf16* __restrict__ dst, int n) {
  int i = blockIdx.x * blockDim.x + threadIdx.x;
  if (i < n) dst[i] = f2bf(src[i]);
}

// 2x2 max pool on bf16 feature maps: [CH, 64, 64] -> [CH, 32, 32]
__global__ void k_maxpool2x2(const __bf16* __restrict__ src,
                             __bf16* __restrict__ dst, int total) {
  int i = blockIdx.x * blockDim.x + threadIdx.x;
  if (i >= total) return;
  int wp = i & 31;
  int hp = (i >> 5) & 31;
  int ch = i >> 10;
  const __bf16* p = src + ((size_t)ch << 12) + ((size_t)(hp << 1) << 6) + (wp << 1);
  float a = bf2f(p[0]),  b = bf2f(p[1]);
  float c = bf2f(p[64]), d = bf2f(p[65]);
  dst[i] = f2bf(fmaxf(fmaxf(a, b), fmaxf(c, d)));
}

// Register-blocked bf16 GEMM: Y[b] = Wt[b?] (OxC) * X[b] (CxN).
// Each wave computes an (MT*16) x (NT*16) output block.
// TRANS=1 stores Y transposed ([N,O] row-major) with packed 16B stores.
template <int MT, int NT, int TRANS>
__global__ void k_gemm_bf16(const __bf16* __restrict__ Wt,
                            const __bf16* __restrict__ X,
                            __bf16* __restrict__ Y,
                            int O, int C, int N,
                            long wStride, long xStride, long yStride) {
  int wave = (int)((blockIdx.x * blockDim.x + threadIdx.x) >> 5);
  int lane = threadIdx.x & 31;
  int NG = N / (16 * NT);
  int OG = O / (16 * MT);
  int tpb = OG * NG;
  int b  = wave / tpb;
  int r  = wave % tpb;
  int og = r / NG, ng = r % NG;
  int row0 = og * (16 * MT);
  int col0 = ng * (16 * NT);

  const __bf16* wb = Wt + (size_t)b * wStride;
  const __bf16* xb = X  + (size_t)b * xStride;

  v8f acc[MT][NT];
#pragma unroll
  for (int i = 0; i < MT; ++i)
#pragma unroll
    for (int j = 0; j < NT; ++j) acc[i][j] = (v8f){};

  for (int k = 0; k < C; k += 32) {
    v16bf a[MT], bb[NT];
#pragma unroll
    for (int i = 0; i < MT; ++i) a[i] = load_frag_a(wb, C, row0 + 16 * i, k);
#pragma unroll
    for (int j = 0; j < NT; ++j) bb[j] = load_frag_b(xb, N, k, col0 + 16 * j);
#pragma unroll
    for (int i = 0; i < MT; ++i)
#pragma unroll
      for (int j = 0; j < NT; ++j)
        acc[i][j] = wmma_bf16(a[i], bb[j], acc[i][j]);
  }

  int m  = lane & 15;
  int hi = (lane >> 4) << 3;
  if (TRANS) {
    __bf16* yb = Y + (size_t)b * yStride;
#pragma unroll
    for (int i = 0; i < MT; ++i)
#pragma unroll
      for (int j = 0; j < NT; ++j) {
        __bf16* yp = yb + (size_t)(col0 + 16 * j + m) * O + row0 + 16 * i + hi;
        v8bf pk;
#pragma unroll
        for (int q = 0; q < 8; ++q) pk[q] = f2bf(acc[i][j][q]);
        *(v8bf*)yp = pk;
      }
  } else {
    __bf16* yb = Y + (size_t)b * yStride;
#pragma unroll
    for (int i = 0; i < MT; ++i)
#pragma unroll
      for (int j = 0; j < NT; ++j)
#pragma unroll
        for (int q = 0; q < 8; ++q)
          yb[(size_t)(row0 + 16 * i + q + hi) * N + col0 + 16 * j + m] =
              f2bf(acc[i][j][q]);
  }
}

// Register-blocked GEMM with double-buffered async global->LDS staging of the
// A panel (shared by all 8 waves of the block). A: [b][O x K], B: [b][K x N],
// Y: [b][O x N]. Block = 8 waves = 8 consecutive n-groups, one o-group.
template <int MT, int NT>
__global__ void k_gemm_async(const __bf16* __restrict__ A,
                             const __bf16* __restrict__ Bm,
                             __bf16* __restrict__ Y,
                             int O, int K, int N,
                             long aStride, long bStride, long yStride) {
  extern __shared__ char lds_raw[];
  const int ROWS  = 16 * MT;                 // A panel rows
  const int PANEL = ROWS * 32 * 2;           // bytes per 32-deep k panel

  int wave = threadIdx.x >> 5;
  int lane = threadIdx.x & 31;
  int NG = N / (16 * NT);
  int OG = O / ROWS;
  int blocksPerBatch = OG * (NG / 8);
  int b   = blockIdx.x / blocksPerBatch;
  int r   = blockIdx.x % blocksPerBatch;
  int og  = r / (NG / 8);
  int ngb = r % (NG / 8);
  int ng  = ngb * 8 + wave;
  int row0 = og * ROWS;
  int col0 = ng * (16 * NT);

  const __bf16* ab = A  + (size_t)b * aStride + (size_t)row0 * K;
  const __bf16* bb = Bm + (size_t)b * bStride;

  // One thread copies 16 B of the ROWSx32 panel: row = t/4, col = (t%4)*8.
  int t    = threadIdx.x;
  int crow = t >> 2;
  int ccol = (t & 3) << 3;
  unsigned lds_elem = (unsigned)(crow * 32 + ccol);

  auto issue_panel = [&](int buf, int k) {
    unsigned dst = (unsigned)(buf * PANEL) + lds_elem * 2u;
    const __bf16* gp = ab + (size_t)crow * K + k + ccol;
    asm volatile("global_load_async_to_lds_b128 %0, %1, off"
                 :: "v"(dst), "v"(gp) : "memory");
  };

  v8f acc[MT][NT];
#pragma unroll
  for (int i = 0; i < MT; ++i)
#pragma unroll
    for (int j = 0; j < NT; ++j) acc[i][j] = (v8f){};

  issue_panel(0, 0);
  int bufc = 0;
  for (int k = 0; k < K; k += 32) {
    asm volatile("s_wait_asynccnt 0" ::: "memory");
    __syncthreads();                         // panel bufc resident for all waves
    if (k + 32 < K) issue_panel(bufc ^ 1, k + 32);

    const __bf16* panel = (const __bf16*)(lds_raw + bufc * PANEL);
    v16bf a[MT], bfr[NT];
#pragma unroll
    for (int i = 0; i < MT; ++i) a[i] = load_frag_a_lds(panel, 16 * i);
#pragma unroll
    for (int j = 0; j < NT; ++j) bfr[j] = load_frag_b(bb, N, k, col0 + 16 * j);
#pragma unroll
    for (int i = 0; i < MT; ++i)
#pragma unroll
      for (int j = 0; j < NT; ++j)
        acc[i][j] = wmma_bf16(a[i], bfr[j], acc[i][j]);

    __syncthreads();                         // done reading bufc before reuse
    bufc ^= 1;
  }

  int m  = lane & 15;
  int hi = (lane >> 4) << 3;
  __bf16* yb = Y + (size_t)b * yStride;
#pragma unroll
  for (int i = 0; i < MT; ++i)
#pragma unroll
    for (int j = 0; j < NT; ++j)
#pragma unroll
      for (int q = 0; q < 8; ++q)
        yb[(size_t)(row0 + 16 * i + q + hi) * N + col0 + 16 * j + m] =
            f2bf(acc[i][j][q]);
}

// Attention scores + softmax + transposed beta store.
// One block = 16 rows (n) x all M=1024 cols. 8 waves x 8 m-tiles, K=64.
__global__ void k_attn_softmax(const __bf16* __restrict__ thetaT, // [B,N,64]
                               const __bf16* __restrict__ phiP,   // [B,64,M]
                               __bf16* __restrict__ betaT,        // [B,M,N]
                               int Nn, int Mm, int Kc) {
  extern __shared__ char smem_raw[];
  const int SLD = 1028;                       // padded row stride (floats)
  float* S      = (float*)smem_raw;           // [16][SLD]
  float* redbuf = S + 16 * SLD;               // [16][16]
  float* rowmax = redbuf + 256;               // [16]
  float* rowsum = rowmax + 16;                // [16]

  int nt16 = Nn >> 4;
  int b  = blockIdx.x / nt16;
  int n0 = (blockIdx.x % nt16) << 4;
  int lane = threadIdx.x & 31;
  int wave = threadIdx.x >> 5;

  const __bf16* tb = thetaT + (size_t)b * Nn * Kc;
  const __bf16* pb = phiP   + (size_t)b * Kc * Mm;

  int m  = lane & 15;
  int hi = (lane >> 4) << 3;

  // theta^T A-fragments are identical for all 8 m-tiles: load once.
  v16bf a0 = load_frag_a(tb, Kc, n0, 0);
  v16bf a1 = load_frag_a(tb, Kc, n0, 32);

#pragma unroll
  for (int j = 0; j < 8; ++j) {
    int mt = (wave << 3) + j;
    v16bf b0 = load_frag_b(pb, Mm, 0,  mt << 4);
    v16bf b1 = load_frag_b(pb, Mm, 32, mt << 4);
    v8f acc = {};
    acc = wmma_bf16(a0, b0, acc);
    acc = wmma_bf16(a1, b1, acc);
#pragma unroll
    for (int q = 0; q < 8; ++q)
      S[(q + hi) * SLD + (mt << 4) + m] = acc[q];
  }
  __syncthreads();

  // softmax over M, 16 threads per row, 64 cols each
  int row = threadIdx.x >> 4;
  int sub = threadIdx.x & 15;
  float* srow = S + row * SLD + (sub << 6);
  float lmax = -3.4e38f;
  for (int i = 0; i < 64; ++i) lmax = fmaxf(lmax, srow[i]);
  redbuf[(row << 4) + sub] = lmax;
  __syncthreads();
  if (sub == 0) {
    float mx = redbuf[row << 4];
    for (int i = 1; i < 16; ++i) mx = fmaxf(mx, redbuf[(row << 4) + i]);
    rowmax[row] = mx;
  }
  __syncthreads();
  float mx = rowmax[row];
  float lsum = 0.f;
  for (int i = 0; i < 64; ++i) {
    float e = __expf(srow[i] - mx);
    srow[i] = e;
    lsum += e;
  }
  redbuf[(row << 4) + sub] = lsum;
  __syncthreads();
  if (sub == 0) {
    float sm = 0.f;
    for (int i = 0; i < 16; ++i) sm += redbuf[(row << 4) + i];
    rowsum[row] = sm;
  }
  __syncthreads();

  float inv[16];
#pragma unroll
  for (int i = 0; i < 16; ++i) inv[i] = 1.0f / rowsum[i];

  // transposed write: betaT[m][n0..n0+15] as one packed 32B store per m
  for (int mm = threadIdx.x; mm < Mm; mm += 256) {
    v16bf pk;
#pragma unroll
    for (int nl = 0; nl < 16; ++nl)
      pk[nl] = f2bf(S[nl * SLD + mm] * inv[nl]);
    *(v16bf*)(betaT + ((size_t)b * Mm + mm) * Nn + n0) = pk;
  }
}

// Final projection + residual blend: Y = g*bal*(Wo @ O) + (1-bal)*X  (fp32 out)
template <int MT, int NT>
__global__ void k_out_blend(const __bf16* __restrict__ Wo,  // [512,256]
                            const __bf16* __restrict__ Ob,  // [B,256,N]
                            const float* __restrict__ Xf,   // [B,512,N]
                            float* __restrict__ Yout,       // [B,512,N]
                            const float* __restrict__ gammaP,
                            const float* __restrict__ balanceP,
                            int O, int C, int N) {
  int wave = (int)((blockIdx.x * blockDim.x + threadIdx.x) >> 5);
  int lane = threadIdx.x & 31;
  int NG = N / (16 * NT);
  int OG = O / (16 * MT);
  int tpb = OG * NG;
  int b  = wave / tpb;
  int r  = wave % tpb;
  int og = r / NG, ng = r % NG;
  int row0 = og * (16 * MT);
  int col0 = ng * (16 * NT);

  const __bf16* ob = Ob + (size_t)b * C * N;
  v8f acc[MT][NT];
#pragma unroll
  for (int i = 0; i < MT; ++i)
#pragma unroll
    for (int j = 0; j < NT; ++j) acc[i][j] = (v8f){};

  for (int k = 0; k < C; k += 32) {
    v16bf a[MT], bb[NT];
#pragma unroll
    for (int i = 0; i < MT; ++i) a[i] = load_frag_a(Wo, C, row0 + 16 * i, k);
#pragma unroll
    for (int j = 0; j < NT; ++j) bb[j] = load_frag_b(ob, N, k, col0 + 16 * j);
#pragma unroll
    for (int i = 0; i < MT; ++i)
#pragma unroll
      for (int j = 0; j < NT; ++j)
        acc[i][j] = wmma_bf16(a[i], bb[j], acc[i][j]);
  }

  float bal = balanceP[0];
  float gb  = gammaP[0] * bal;
  float rb  = 1.0f - bal;
  int m  = lane & 15;
  int hi = (lane >> 4) << 3;
  size_t base = (size_t)b * O * N;
#pragma unroll
  for (int i = 0; i < MT; ++i)
#pragma unroll
    for (int j = 0; j < NT; ++j)
#pragma unroll
      for (int q = 0; q < 8; ++q) {
        size_t idx = base + (size_t)(row0 + 16 * i + q + hi) * N +
                     col0 + 16 * j + m;
        Yout[idx] = gb * acc[i][j][q] + rb * Xf[idx];
      }
}

// ---------------------------------------------------------------------------
extern "C" void kernel_launch(void* const* d_in, const int* in_sizes, int n_in,
                              void* d_out, int out_size, void* d_ws, size_t ws_size,
                              hipStream_t stream) {
  const float* x        = (const float*)d_in[0];
  const float* w_theta  = (const float*)d_in[1];
  const float* w_phi    = (const float*)d_in[2];
  const float* w_g      = (const float*)d_in[3];
  const float* w_o      = (const float*)d_in[4];
  const float* gamma    = (const float*)d_in[5];
  const float* balance  = (const float*)d_in[6];
  float* out = (float*)d_out;

  const int B = 8, C = 512, N = 4096, M = 1024, C8 = 64, C2 = 256;

  char* ws = (char*)d_ws;
  size_t off = 0;
  auto carve = [&](size_t bytes) -> char* {
    char* p = ws + off;
    off += (bytes + 255) & ~(size_t)255;
    return p;
  };
  __bf16* x_bf   = (__bf16*)carve((size_t)B * C * N * 2);   // 32 MB
  __bf16* thetaT = (__bf16*)carve((size_t)B * N * C8 * 2);  //  4 MB  [B,N,64]
  __bf16* phi_un = (__bf16*)carve((size_t)B * C8 * N * 2);  //  4 MB
  __bf16* g_un   = (__bf16*)carve((size_t)B * C2 * N * 2);  // 16 MB
  __bf16* phi_p  = (__bf16*)carve((size_t)B * C8 * M * 2);  //  1 MB
  __bf16* g_p    = (__bf16*)carve((size_t)B * C2 * M * 2);  //  4 MB
  __bf16* betaT  = (__bf16*)carve((size_t)B * M * N * 2);   // 64 MB  [B,M,N]
  __bf16* o_bf   = (__bf16*)carve((size_t)B * C2 * N * 2);  // 16 MB
  __bf16* wth_bf = (__bf16*)carve((size_t)C8 * C * 2);
  __bf16* wph_bf = (__bf16*)carve((size_t)C8 * C * 2);
  __bf16* wg_bf  = (__bf16*)carve((size_t)C2 * C * 2);
  __bf16* wo_bf  = (__bf16*)carve((size_t)C * C2 * 2);

  // fp32 -> bf16 converts (activations + weights); everything stays in L2.
  {
    int n = B * C * N;
    k_f32_to_bf16<<<(n + 255) / 256, 256, 0, stream>>>(x, x_bf, n);
    n = C8 * C;
    k_f32_to_bf16<<<(n + 255) / 256, 256, 0, stream>>>(w_theta, wth_bf, n);
    k_f32_to_bf16<<<(n + 255) / 256, 256, 0, stream>>>(w_phi, wph_bf, n);
    n = C2 * C;
    k_f32_to_bf16<<<(n + 255) / 256, 256, 0, stream>>>(w_g, wg_bf, n);
    n = C * C2;
    k_f32_to_bf16<<<(n + 255) / 256, 256, 0, stream>>>(w_o, wo_bf, n);
  }
  // 1x1-conv projections (channel GEMMs), K = 512, 32x64 per wave
  {
    int waves = B * (C8 / 32) * (N / 64);                   // 1024
    k_gemm_bf16<2, 4, 1><<<waves / 8, 256, 0, stream>>>(
        wth_bf, x_bf, thetaT, C8, C, N, 0L, (long)C * N, (long)N * C8);
    k_gemm_bf16<2, 4, 0><<<waves / 8, 256, 0, stream>>>(
        wph_bf, x_bf, phi_un, C8, C, N, 0L, (long)C * N, (long)C8 * N);
    waves = B * (C2 / 32) * (N / 64);                       // 4096
    k_gemm_bf16<2, 4, 0><<<waves / 8, 256, 0, stream>>>(
        wg_bf, x_bf, g_un, C2, C, N, 0L, (long)C * N, (long)C2 * N);
  }
  // 2x2 max pooling for phi and g
  {
    int n = B * C8 * M;
    k_maxpool2x2<<<(n + 255) / 256, 256, 0, stream>>>(phi_un, phi_p, n);
    n = B * C2 * M;
    k_maxpool2x2<<<(n + 255) / 256, 256, 0, stream>>>(g_un, g_p, n);
  }
  // attention scores + softmax, writes beta transposed [B,M,N]
  {
    size_t lds = (size_t)(16 * 1028 + 256 + 32) * sizeof(float); // ~66.9 KB
    k_attn_softmax<<<B * (N / 16), 256, lds, stream>>>(thetaT, phi_p, betaT, N, M, C8);
  }
  // o = g_p (256x1024, per batch) @ betaT (1024xN), K = 1024, 64x64 per wave,
  // A panel staged through LDS with double-buffered async copies.
  {
    int blocks = B * (C2 / 64) * ((N / 64) / 8);            // 256
    size_t lds = 2 * (64 * 32 * 2);                         // 8 KB
    k_gemm_async<4, 4><<<blocks, 256, lds, stream>>>(
        g_p, betaT, o_bf, C2, M, N, (long)C2 * M, (long)M * N, (long)C2 * N);
  }
  // final projection + residual blend, K = 256, fp32 output, 64x64 per wave
  {
    int waves = B * (C / 64) * (N / 64);                    // 4096
    k_out_blend<4, 4><<<waves / 8, 256, 0, stream>>>(
        wo_bf, o_bf, x, out, gamma, balance, C, C2, N);
  }
}
